// GINConvNet_63402307224307
// MI455X (gfx1250) — matrix operand order repaired and checked by reading, not compiled, not run on hardware
//
#include <hip/hip_runtime.h>
#include <hip/hip_bf16.h>

// ---------------------------------------------------------------------------
// CDNA5 (gfx1250) GIN forward. wave32, WMMA bf16 (f32 accumulate).
// ---------------------------------------------------------------------------

typedef __attribute__((ext_vector_type(16))) __bf16 v16bf;
typedef __attribute__((ext_vector_type(8)))  float  v8f;
typedef __attribute__((ext_vector_type(4)))  float  f4;

#define N_NODES 50000
#define N_EDGES 800000
#define N_GRAPHS 256
#define FIN 112
#define DIM 32

__device__ __forceinline__ v8f vzero8() {
    v8f v;
#pragma unroll
    for (int i = 0; i < 8; ++i) v[i] = 0.0f;
    return v;
}

// ---------------------------------------------------------------------------
// A-fragment (16x32 bf16, ISA 7.12.2 layout). Element j in [0,8) holds
// k = kc*32 + 8*half + j ; element 8+j holds k = kc*32 + 16 + 8*half + j.
// Both runs are contiguous 8-float spans -> 2x b128 loads per source.
// Tail validity (kc*32+31 < kmax) is half-independent and compile-time
// constant once the kc loop is unrolled.
// ---------------------------------------------------------------------------
__device__ __forceinline__ v16bf load_a_sum(const float* __restrict__ A,
                                            const float* __restrict__ B,
                                            long rowbase, int kc, int kmax,
                                            int half) {
    v16bf a;
    const long p0 = rowbase + kc * 32 + 8 * half;
    f4 x0 = *(const f4*)(A + p0);
    f4 x1 = *(const f4*)(A + p0 + 4);
    f4 y0 = *(const f4*)(B + p0);
    f4 y1 = *(const f4*)(B + p0 + 4);
#pragma unroll
    for (int j = 0; j < 4; ++j) {
        a[j]     = (__bf16)(x0[j] + y0[j]);
        a[4 + j] = (__bf16)(x1[j] + y1[j]);
    }
    if (kc * 32 + 31 < kmax) {
        f4 x2 = *(const f4*)(A + p0 + 16);
        f4 x3 = *(const f4*)(A + p0 + 20);
        f4 y2 = *(const f4*)(B + p0 + 16);
        f4 y3 = *(const f4*)(B + p0 + 20);
#pragma unroll
        for (int j = 0; j < 4; ++j) {
            a[8 + j]  = (__bf16)(x2[j] + y2[j]);
            a[12 + j] = (__bf16)(x3[j] + y3[j]);
        }
    } else {
#pragma unroll
        for (int j = 0; j < 8; ++j) a[8 + j] = (__bf16)0.0f;
    }
    return a;
}

__device__ __forceinline__ v16bf load_a_lds(const float* L, int stride, int row,
                                            int kc, int kmax, int half) {
    v16bf a;
    const int p0 = row * stride + kc * 32 + 8 * half;
    f4 x0 = *(const f4*)(L + p0);
    f4 x1 = *(const f4*)(L + p0 + 4);
#pragma unroll
    for (int j = 0; j < 4; ++j) {
        a[j]     = (__bf16)x0[j];
        a[4 + j] = (__bf16)x1[j];
    }
    if (kc * 32 + 31 < kmax) {
        f4 x2 = *(const f4*)(L + p0 + 16);
        f4 x3 = *(const f4*)(L + p0 + 20);
#pragma unroll
        for (int j = 0; j < 4; ++j) {
            a[8 + j]  = (__bf16)x2[j];
            a[12 + j] = (__bf16)x3[j];
        }
    } else {
#pragma unroll
        for (int j = 0; j < 8; ++j) a[8 + j] = (__bf16)0.0f;
    }
    return a;
}

// ---------------------------------------------------------------------------
__global__ void zero_kernel(float* p, long n) {
    long i = (long)blockIdx.x * blockDim.x + threadIdx.x;
    if (i < n) p[i] = 0.0f;
}

// Pack W[K x N] (row-major) into per-lane B-fragments (32x16 bf16 per wmma).
// out index: ((ct*KC + kc)*32 + lane) * 16 bf16 elems.
__global__ void pack_frags_kernel(const float* __restrict__ W, int K, int N,
                                  int CT, int KC, __bf16* __restrict__ out) {
    int idx = blockIdx.x * blockDim.x + threadIdx.x;
    int total = CT * KC * 32;
    if (idx >= total) return;
    int lane = idx & 31;
    int fk   = idx >> 5;
    int kc   = fk % KC;
    int ct   = fk / KC;
    int half = lane >> 4;
    int n    = ct * 16 + (lane & 15);
    v16bf v;
#pragma unroll
    for (int i = 0; i < 8; ++i) {
        int k = kc * 32 + 2 * i + 8 * half + (i >= 4 ? 8 : 0);  // B 32x16: elem s<8 K=8*half+s, s>=8 K=16+8*half+(s-8)
        float f0 = (k     < K && n < N) ? W[(long)k * N + n]       : 0.f;
        float f1 = (k + 1 < K && n < N) ? W[(long)(k + 1) * N + n] : 0.f;
        v[2 * i]     = (__bf16)f0;
        v[2 * i + 1] = (__bf16)f1;
    }
    *(v16bf*)(out + (long)idx * 16) = v;
}

// agg[dst[e]] += feat[src[e]]  (width 112)
__global__ __launch_bounds__(128) void edge_agg_kernel(
    const float* __restrict__ feat, const int* __restrict__ src,
    const int* __restrict__ dst, float* __restrict__ agg) {
    int e = blockIdx.x;
    int f = threadIdx.x;
    if (f >= FIN) return;
    int s = src[e], d = dst[e];
    atomicAdd(&agg[(long)d * FIN + f], feat[(long)s * FIN + f]);
}

// ---------------------------------------------------------------------------
// Fused GIN layer 1 MLP: h1 = relu(relu((x+agg)@W1 + b1)@W2 + b2), + BN stats.
// 4 waves / block, wave = one 16-row node tile. 7 col-tiles, 4 k-chunks.
// N_NODES % 16 == 0, so every in-range tile has 16 valid rows; only the
// padding tiles (row0 >= N_NODES) clamp loads and skip the epilogue.
// ---------------------------------------------------------------------------
__global__ __launch_bounds__(128) void mlp1_kernel(
    const float* __restrict__ x, const float* __restrict__ agg,
    const __bf16* __restrict__ fw1, const __bf16* __restrict__ fw2,
    const float* __restrict__ b1, const float* __restrict__ b2,
    float* __restrict__ h1, float* __restrict__ stats) {
    __shared__ float lds[4][16 * 136];
    const int wave = threadIdx.x >> 5;
    const int lane = threadIdx.x & 31;
    const int half = lane >> 4;
    const int l16  = lane & 15;
    const int row0 = (blockIdx.x * 4 + wave) * 16;
    const int lrow = min(row0 + l16, N_NODES - 1);   // clamp for padding tiles
    const long rb  = (long)lrow * FIN;

    v8f acc[7];
#pragma unroll
    for (int ct = 0; ct < 7; ++ct) acc[ct] = vzero8();

    // GEMM 1a: (x+agg)[16x112] @ W1[112x112]
#pragma unroll
    for (int kc = 0; kc < 4; ++kc) {
        v16bf a = load_a_sum(x, agg, rb, kc, FIN, half);
#pragma unroll
        for (int ct = 0; ct < 7; ++ct) {
            v16bf b = *(const v16bf*)(fw1 + ((long)(ct * 4 + kc) * 32 + lane) * 16);
            acc[ct] = __builtin_amdgcn_wmma_f32_16x16x32_bf16(
                false, a, false, b, (short)0, acc[ct], false, false);
        }
    }
    // bias + relu -> LDS tile (16 x 112, stride 136), reset accumulators
    float* L = &lds[wave][0];
#pragma unroll
    for (int ct = 0; ct < 7; ++ct) {
        float bias = b1[ct * 16 + l16];
#pragma unroll
        for (int r = 0; r < 8; ++r)
            L[(r + 8 * half) * 136 + ct * 16 + l16] = fmaxf(acc[ct][r] + bias, 0.f);
        acc[ct] = vzero8();
    }
    __syncthreads();

    // GEMM 1b: t[16x112] @ W2[112x112]
#pragma unroll
    for (int kc = 0; kc < 4; ++kc) {
        v16bf a = load_a_lds(L, 136, l16, kc, FIN, half);
#pragma unroll
        for (int ct = 0; ct < 7; ++ct) {
            v16bf b = *(const v16bf*)(fw2 + ((long)(ct * 4 + kc) * 32 + lane) * 16);
            acc[ct] = __builtin_amdgcn_wmma_f32_16x16x32_bf16(
                false, a, false, b, (short)0, acc[ct], false, false);
        }
    }
    // epilogue: bias, relu, store h1, BN1 partial sums (whole tile valid)
    if (row0 < N_NODES) {
#pragma unroll
        for (int ct = 0; ct < 7; ++ct) {
            float bias = b2[ct * 16 + l16];
            float s = 0.f, sq = 0.f;
#pragma unroll
            for (int r = 0; r < 8; ++r) {
                int gr = row0 + r + 8 * half;
                float v = fmaxf(acc[ct][r] + bias, 0.f);
                h1[(long)gr * FIN + ct * 16 + l16] = v;
                s += v; sq += v * v;
            }
            atomicAdd(&stats[ct * 16 + l16], s);
            atomicAdd(&stats[FIN + ct * 16 + l16], sq);
        }
    }
}

// ---------------------------------------------------------------------------
// Fused GIN layer 2 MLP: h2 = relu(relu((h+agg)@W1 + b1)@W2 + b2), + BN stats.
// ---------------------------------------------------------------------------
__global__ __launch_bounds__(128) void mlp2_kernel(
    const float* __restrict__ h, const float* __restrict__ agg,
    const __bf16* __restrict__ fw1, const __bf16* __restrict__ fw2,
    const float* __restrict__ b1, const float* __restrict__ b2,
    float* __restrict__ h2, float* __restrict__ stats) {
    __shared__ float lds[4][16 * 40];
    const int wave = threadIdx.x >> 5;
    const int lane = threadIdx.x & 31;
    const int half = lane >> 4;
    const int l16  = lane & 15;
    const int row0 = (blockIdx.x * 4 + wave) * 16;
    const int lrow = min(row0 + l16, N_NODES - 1);
    const long rb  = (long)lrow * FIN;

    v8f acc[2];
    acc[0] = vzero8(); acc[1] = vzero8();

    // GEMM 2a: (h+agg)[16x112] @ W1[112x32]
#pragma unroll
    for (int kc = 0; kc < 4; ++kc) {
        v16bf a = load_a_sum(h, agg, rb, kc, FIN, half);
#pragma unroll
        for (int ct = 0; ct < 2; ++ct) {
            v16bf b = *(const v16bf*)(fw1 + ((long)(ct * 4 + kc) * 32 + lane) * 16);
            acc[ct] = __builtin_amdgcn_wmma_f32_16x16x32_bf16(
                false, a, false, b, (short)0, acc[ct], false, false);
        }
    }
    float* L = &lds[wave][0];
#pragma unroll
    for (int ct = 0; ct < 2; ++ct) {
        float bias = b1[ct * 16 + l16];
#pragma unroll
        for (int r = 0; r < 8; ++r)
            L[(r + 8 * half) * 40 + ct * 16 + l16] = fmaxf(acc[ct][r] + bias, 0.f);
        acc[ct] = vzero8();
    }
    __syncthreads();

    // GEMM 2b: t[16x32] @ W2[32x32]  (single K-chunk, fully in-range)
    {
        v16bf a = load_a_lds(L, 40, l16, 0, DIM, half);
#pragma unroll
        for (int ct = 0; ct < 2; ++ct) {
            v16bf b = *(const v16bf*)(fw2 + ((long)ct * 32 + lane) * 16);
            acc[ct] = __builtin_amdgcn_wmma_f32_16x16x32_bf16(
                false, a, false, b, (short)0, acc[ct], false, false);
        }
    }
    if (row0 < N_NODES) {
#pragma unroll
        for (int ct = 0; ct < 2; ++ct) {
            float bias = b2[ct * 16 + l16];
            float s = 0.f, sq = 0.f;
#pragma unroll
            for (int r = 0; r < 8; ++r) {
                int gr = row0 + r + 8 * half;
                float v = fmaxf(acc[ct][r] + bias, 0.f);
                h2[(long)gr * DIM + ct * 16 + l16] = v;
                s += v; sq += v * v;
            }
            atomicAdd(&stats[ct * 16 + l16], s);
            atomicAdd(&stats[DIM + ct * 16 + l16], sq);
        }
    }
}

// ---------------------------------------------------------------------------
__global__ void bn_finalize_kernel(const float* __restrict__ stats,
                                   const float* __restrict__ gamma,
                                   const float* __restrict__ beta,
                                   float* __restrict__ scale,
                                   float* __restrict__ shift, int F, float invN) {
    int f = threadIdx.x;
    if (f >= F) return;
    float m = stats[f] * invN;
    float v = stats[F + f] * invN - m * m;
    float sc = gamma[f] * rsqrtf(v + 1e-5f);
    scale[f] = sc;
    shift[f] = beta[f] - m * sc;
}

__global__ void bn_apply_kernel(float* __restrict__ h, const float* __restrict__ scale,
                                const float* __restrict__ shift, int n) {
    int i = blockIdx.x * blockDim.x + threadIdx.x;
    if (i >= n) return;
    int f = i % FIN;
    h[i] = h[i] * scale[f] + shift[f];
}

// BN2 apply + global_add_pool into pooled[256 x 32]
__global__ void bn2_pool_kernel(const float* __restrict__ h2,
                                const float* __restrict__ scale,
                                const float* __restrict__ shift,
                                const int* __restrict__ batch,
                                float* __restrict__ pooled, int n) {
    int i = blockIdx.x * blockDim.x + threadIdx.x;
    if (i >= n) return;
    int node = i >> 5, f = i & 31;
    float v = h2[i] * scale[f] + shift[f];
    atomicAdd(&pooled[(long)batch[node] * DIM + f], v);
}

// Dense head: one thread per graph.
__global__ __launch_bounds__(256) void head_kernel(
    const float* __restrict__ pooled,
    const float* __restrict__ fxw, const float* __restrict__ fxb,
    const float* __restrict__ w1, const float* __restrict__ bb1,
    const float* __restrict__ w2, const float* __restrict__ bb2,
    const float* __restrict__ w3, const float* __restrict__ bb3,
    const float* __restrict__ w4, const float* __restrict__ bb4,
    const float* __restrict__ w5, const float* __restrict__ bb5,
    float* __restrict__ out) {
    int g = threadIdx.x;
    if (g >= N_GRAPHS) return;
    const float* p = pooled + (long)g * DIM;
    float z[64];
    for (int j = 0; j < 64; ++j) {
        float s = fxb[j];
        for (int i = 0; i < 32; ++i) s += p[i] * fxw[i * 64 + j];
        z[j] = fmaxf(s, 0.f);
    }
    float a1[32];
    for (int j = 0; j < 32; ++j) {
        float s = bb1[j];
        for (int i = 0; i < 64; ++i) s += z[i] * w1[i * 32 + j];
        a1[j] = s;
    }
    float a2[16];
    for (int j = 0; j < 16; ++j) {
        float s = bb2[j];
        for (int i = 0; i < 32; ++i) s += a1[i] * w2[i * 16 + j];
        a2[j] = s;
    }
    float a3[8];
    for (int j = 0; j < 8; ++j) {
        float s = bb3[j];
        for (int i = 0; i < 16; ++i) s += a2[i] * w3[i * 8 + j];
        a3[j] = s;
    }
    float a4[2];
    for (int j = 0; j < 2; ++j) {
        float s = bb4[j];
        for (int i = 0; i < 8; ++i) s += a3[i] * w4[i * 2 + j];
        a4[j] = s;
    }
    out[g] = bb5[0] + a4[0] * w5[0] + a4[1] * w5[1];
}

// ---------------------------------------------------------------------------
extern "C" void kernel_launch(void* const* d_in, const int* in_sizes, int n_in,
                              void* d_out, int out_size, void* d_ws, size_t ws_size,
                              hipStream_t stream) {
    const float* x      = (const float*)d_in[0];
    const int*   src    = (const int*)  d_in[1];
    const int*   dst    = (const int*)  d_in[2];
    const int*   batch  = (const int*)  d_in[3];
    const float* g1_w1  = (const float*)d_in[4];
    const float* g1_b1  = (const float*)d_in[5];
    const float* g1_w2  = (const float*)d_in[6];
    const float* g1_b2  = (const float*)d_in[7];
    const float* bn1_g  = (const float*)d_in[8];
    const float* bn1_b  = (const float*)d_in[9];
    const float* g2_w1  = (const float*)d_in[10];
    const float* g2_b1  = (const float*)d_in[11];
    const float* g2_w2  = (const float*)d_in[12];
    const float* g2_b2  = (const float*)d_in[13];
    const float* bn2_g  = (const float*)d_in[14];
    const float* bn2_b  = (const float*)d_in[15];
    const float* fcxd_w = (const float*)d_in[16];
    const float* fcxd_b = (const float*)d_in[17];
    const float* fc1_w  = (const float*)d_in[18];
    const float* fc1_b  = (const float*)d_in[19];
    const float* fc2_w  = (const float*)d_in[20];
    const float* fc2_b  = (const float*)d_in[21];
    const float* fc3_w  = (const float*)d_in[22];
    const float* fc3_b  = (const float*)d_in[23];
    const float* fc4_w  = (const float*)d_in[24];
    const float* fc4_b  = (const float*)d_in[25];
    const float* fc5_w  = (const float*)d_in[26];
    const float* fc5_b  = (const float*)d_in[27];
    float* out = (float*)d_out;

    float* ws = (float*)d_ws;
    const long NFIN = (long)N_NODES * FIN;       // 5,600,000
    float* h1     = ws;                           // [50000 x 112]
    float* agg    = ws + NFIN;                    // [50000 x 112] (reused both layers)
    float* h2     = ws + 2 * NFIN;                // [50000 x 32]
    float* stats1 = h2 + (long)N_NODES * DIM;     // 224
    float* stats2 = stats1 + 2 * FIN;             // 64
    float* bn1_sc = stats2 + 2 * DIM;             // 112
    float* bn1_sh = bn1_sc + FIN;                 // 112
    float* bn2_sc = bn1_sh + FIN;                 // 32
    float* bn2_sh = bn2_sc + DIM;                 // 32
    float* pooled = bn2_sh + DIM;                 // [256 x 32]
    __bf16* frags = (__bf16*)(pooled + (long)N_GRAPHS * DIM);
    __bf16* fw11 = frags;                         // 7*4*32*16
    __bf16* fw12 = fw11 + 7 * 4 * 32 * 16;
    __bf16* fw21 = fw12 + 7 * 4 * 32 * 16;       // 2*4*32*16
    __bf16* fw22 = fw21 + 2 * 4 * 32 * 16;       // 2*1*32*16

    const int ZB = 256;
    long zsmall = 2 * FIN + 2 * DIM + 2 * FIN + 2 * DIM + (long)N_GRAPHS * DIM;

    // --- zero scratch ---
    zero_kernel<<<(int)((NFIN + ZB - 1) / ZB), ZB, 0, stream>>>(agg, NFIN);
    zero_kernel<<<(int)((zsmall + ZB - 1) / ZB), ZB, 0, stream>>>(stats1, zsmall);

    // --- pack weight fragments (bf16 B-layout) ---
    pack_frags_kernel<<<(7 * 4 * 32 + 63) / 64, 64, 0, stream>>>(g1_w1, FIN, FIN, 7, 4, fw11);
    pack_frags_kernel<<<(7 * 4 * 32 + 63) / 64, 64, 0, stream>>>(g1_w2, FIN, FIN, 7, 4, fw12);
    pack_frags_kernel<<<(2 * 4 * 32 + 63) / 64, 64, 0, stream>>>(g2_w1, FIN, DIM, 2, 4, fw21);
    pack_frags_kernel<<<(2 * 1 * 32 + 63) / 64, 64, 0, stream>>>(g2_w2, DIM, DIM, 2, 1, fw22);

    // --- layer 1 ---
    edge_agg_kernel<<<N_EDGES, 128, 0, stream>>>(x, src, dst, agg);
    const int TILES = (N_NODES + 15) / 16;               // 3125
    const int MB = (TILES + 3) / 4;                      // 782 blocks of 4 waves
    mlp1_kernel<<<MB, 128, 0, stream>>>(x, agg, fw11, fw12, g1_b1, g1_b2, h1, stats1);
    bn_finalize_kernel<<<1, 128, 0, stream>>>(stats1, bn1_g, bn1_b, bn1_sc, bn1_sh,
                                              FIN, 1.0f / (float)N_NODES);
    bn_apply_kernel<<<(int)((NFIN + ZB - 1) / ZB), ZB, 0, stream>>>(h1, bn1_sc, bn1_sh, (int)NFIN);

    // --- layer 2 ---
    zero_kernel<<<(int)((NFIN + ZB - 1) / ZB), ZB, 0, stream>>>(agg, NFIN);
    edge_agg_kernel<<<N_EDGES, 128, 0, stream>>>(h1, src, dst, agg);
    mlp2_kernel<<<MB, 128, 0, stream>>>(h1, agg, fw21, fw22, g2_b1, g2_b2, h2, stats2);
    bn_finalize_kernel<<<1, 128, 0, stream>>>(stats2, bn2_g, bn2_b, bn2_sc, bn2_sh,
                                              DIM, 1.0f / (float)N_NODES);

    // --- pool + head ---
    const long NH2 = (long)N_NODES * DIM;
    bn2_pool_kernel<<<(int)((NH2 + ZB - 1) / ZB), ZB, 0, stream>>>(h2, bn2_sc, bn2_sh,
                                                                   batch, pooled, (int)NH2);
    head_kernel<<<1, 256, 0, stream>>>(pooled, fcxd_w, fcxd_b, fc1_w, fc1_b, fc2_w, fc2_b,
                                       fc3_w, fc3_b, fc4_w, fc4_b, fc5_w, fc5_b, out);

    (void)in_sizes; (void)n_in; (void)out_size; (void)ws_size;
}
